// EMAVectorQuantizer_48206712930872
// MI455X (gfx1250) — compile-verified, matrix operand order checked by reading, not run on hardware
//
#include <hip/hip_runtime.h>
#include <hip/hip_bf16.h>

typedef _Float16 v16h __attribute__((ext_vector_type(16)));
typedef _Float16 v8h  __attribute__((ext_vector_type(8)));
typedef float    v8f  __attribute__((ext_vector_type(8)));

#define KCODES 8192
#define DDIM   256
#define BATCH  8
#define HH     32
#define WW     32
#define HW     1024        // HH*WW
#define DHW    262144      // DDIM*HW
#define NVEC   8192        // BATCH*HW
#define TOT    2097152     // NVEC*DDIM
#define DECAY  0.99f
#define ONE_M_DECAY 0.01f
#define BETA   0.25f
#define EPSV   1e-05f

#define NSEG   4           // codebook split for occupancy
#define SEGLEN (KCODES / NSEG)   // 2048 codes per segment

// ---------------- workspace layout (bytes) ----------------
#define WS_ZF16   0            // [NVEC*DDIM] half   (4,194,304)
#define WS_EF16   4194304      // [KCODES*DDIM] half (4,194,304)
#define WS_DW     8388608      // [KCODES*DDIM] f32  (8,388,608)
#define WS_ENORM  16777216     // [KCODES] f32       (32,768)
#define WS_COUNTS 16809984     // [KCODES] f32       (32,768)
#define WS_IDXI   16842752     // [NVEC] int         (32,768)
#define WS_NSUM   16875520     // [1] f32 (+pad)
#define WS_BESTV  16875648     // [NSEG*NVEC] f32    (131,072)
#define WS_BESTI  17006720     // [NSEG*NVEC] int    (131,072)

// ---------------- output layout (floats) ----------------
#define O_ZQ    0           // 2,097,152
#define O_LOSS  2097152     // 1
#define O_IDX   2097153     // 8,192
#define O_CS    2105345     // 8,192
#define O_EMAW  2113537     // 2,097,152
#define O_EMB   4210689     // 2,097,152

// -------- init: zero dw, counts, n_sum, loss slot --------
__global__ void vq_init_kernel(float* __restrict__ dw, float* __restrict__ counts,
                               float* __restrict__ n_sum, float* __restrict__ loss) {
    int t = blockIdx.x * blockDim.x + threadIdx.x;
    if (t < KCODES * DDIM) dw[t] = 0.0f;
    if (t < KCODES) counts[t] = 0.0f;
    if (t == 0) { n_sum[0] = 0.0f; loss[0] = 0.0f; }
}

// -------- prep z: NCHW f32 -> [N,D] row-major f16 --------
__global__ void vq_prep_z_kernel(const float* __restrict__ z, _Float16* __restrict__ zf16) {
    int t = blockIdx.x * blockDim.x + threadIdx.x;   // coalesced read of z (NCHW linear)
    if (t >= TOT) return;
    int b  = t / DHW;
    int d  = (t / HW) % DDIM;
    int hw = t % HW;
    int n  = b * HW + hw;
    zf16[(size_t)n * DDIM + d] = (_Float16)z[t];
}

// -------- prep e: f32 -> f16 copy + row norms --------
__global__ __launch_bounds__(256)
void vq_prep_e_kernel(const float* __restrict__ emb, _Float16* __restrict__ ef16,
                      float* __restrict__ enorm) {
    int k = blockIdx.x;           // one block per code
    int d = threadIdx.x;
    float v = emb[(size_t)k * DDIM + d];
    ef16[(size_t)k * DDIM + d] = (_Float16)v;
    __shared__ float s[256];
    s[d] = v * v;
    __syncthreads();
    for (int off = 128; off > 0; off >>= 1) {
        if (d < off) s[d] += s[d + off];
        __syncthreads();
    }
    if (d == 0) enorm[k] = s[0];
}

// -------- WMMA argmin: 32 rows x 1 codebook segment per wave32 workgroup --------
// A 16-bit 16x32 fragment layout (ISA 7.12.2): lane<16 holds K 0-7 (v0-3) and
// 16-23 (v4-7); lane>=16 holds K 8-15 and 24-31. B (32x16 = e^T chunk) is the
// transpose-symmetric layout, so the same per-lane loader serves both.
// Grid: (NVEC/32) x NSEG -> 1024 waves for occupancy / latency hiding.
__global__ __launch_bounds__(32)
void vq_argmin_wmma_kernel(const _Float16* __restrict__ zf16,
                           const _Float16* __restrict__ ef16,
                           const float* __restrict__ enorm,
                           float* __restrict__ bestv,   // [NSEG*NVEC]
                           int* __restrict__ besti) {   // [NSEG*NVEC]
    const int lane = threadIdx.x;          // 0..31
    const int half_sel = lane >> 4;        // K-subset select
    const int sub = lane & 15;
    const int rowBase = blockIdx.x * 32;
    const int seg = blockIdx.y;
    const int cStart = seg * SEGLEN;
    const int cEnd   = cStart + SEGLEN;

    // Two A-tile sets (rows rowBase..+15 and rowBase+16..+31), resident in VGPRs.
    v16h a0[8], a1[8];
    {
        const _Float16* z0 = zf16 + (size_t)(rowBase + sub) * DDIM;
        const _Float16* z1 = z0 + (size_t)16 * DDIM;
        #pragma unroll
        for (int f = 0; f < 8; ++f) {
            const int kb0 = f * 32 + half_sel * 8;
            v8h lo0 = *(const v8h*)(z0 + kb0);
            v8h hi0 = *(const v8h*)(z0 + kb0 + 16);
            v8h lo1 = *(const v8h*)(z1 + kb0);
            v8h hi1 = *(const v8h*)(z1 + kb0 + 16);
            #pragma unroll
            for (int j = 0; j < 8; ++j) {
                a0[f][j] = lo0[j]; a0[f][8 + j] = hi0[j];
                a1[f][j] = lo1[j]; a1[f][8 + j] = hi1[j];
            }
        }
    }

    float best0[8], best1[8];
    int bidx0[8], bidx1[8];
    #pragma unroll
    for (int r = 0; r < 8; ++r) {
        best0[r] = 3.4e38f; bidx0[r] = 0;
        best1[r] = 3.4e38f; bidx1[r] = 0;
    }

    for (int cBase = cStart; cBase < cEnd; cBase += 16) {
        const int code = cBase + sub;
        const _Float16* esrc = ef16 + (size_t)code * DDIM;
        const float en = enorm[code];                 // hoisted off the critical path
        __builtin_prefetch(esrc + 16 * DDIM, 0, 0);   // next tile -> global_prefetch_b8

        v16h bfr[8];
        #pragma unroll
        for (int f = 0; f < 8; ++f) {
            const int kb0 = f * 32 + half_sel * 8;
            v8h lo = *(const v8h*)(esrc + kb0);
            v8h hi = *(const v8h*)(esrc + kb0 + 16);
            #pragma unroll
            for (int j = 0; j < 8; ++j) { bfr[f][j] = lo[j]; bfr[f][8 + j] = hi[j]; }
        }

        // Interleaved dual accumulator chains: dep distance 2 per chain.
        v8f acc0 = {}, acc1 = {};
        #pragma unroll
        for (int f = 0; f < 8; ++f) {
            acc0 = __builtin_amdgcn_wmma_f32_16x16x32_f16(
                false, a0[f], false, bfr[f], (short)0, acc0, false, false);
            acc1 = __builtin_amdgcn_wmma_f32_16x16x32_f16(
                false, a1[f], false, bfr[f], (short)0, acc1, false, false);
        }

        #pragma unroll
        for (int r = 0; r < 8; ++r) {
            float d0 = en - 2.0f * acc0[r];   // ||z||^2 constant per row: dropped
            float d1 = en - 2.0f * acc1[r];
            if (d0 < best0[r]) { best0[r] = d0; bidx0[r] = code; }
            if (d1 < best1[r]) { best1[r] = d1; bidx1[r] = code; }
        }
    }

    // min-reduce across the 16 lanes of each half (C layout: lane = code column)
    #pragma unroll
    for (int off = 8; off >= 1; off >>= 1) {
        #pragma unroll
        for (int r = 0; r < 8; ++r) {
            float ov0 = __shfl_xor(best0[r], off, 32);
            int   oi0 = __shfl_xor(bidx0[r], off, 32);
            if (ov0 < best0[r] || (ov0 == best0[r] && oi0 < bidx0[r])) {
                best0[r] = ov0; bidx0[r] = oi0;
            }
            float ov1 = __shfl_xor(best1[r], off, 32);
            int   oi1 = __shfl_xor(bidx1[r], off, 32);
            if (ov1 < best1[r] || (ov1 == best1[r] && oi1 < bidx1[r])) {
                best1[r] = ov1; bidx1[r] = oi1;
            }
        }
    }
    if (sub == 0) {
        #pragma unroll
        for (int r = 0; r < 8; ++r) {
            int row0 = rowBase + half_sel * 8 + r;        // tile0: VGPR r = M r+8*half
            bestv[(size_t)seg * NVEC + row0] = best0[r];
            besti[(size_t)seg * NVEC + row0] = bidx0[r];
            int row1 = row0 + 16;                          // tile1
            bestv[(size_t)seg * NVEC + row1] = best1[r];
            besti[(size_t)seg * NVEC + row1] = bidx1[r];
        }
    }
}

// -------- reduce NSEG partial argmins per row --------
__global__ void vq_argmin_final_kernel(const float* __restrict__ bestv,
                                       const int* __restrict__ besti,
                                       float* __restrict__ idx_f,
                                       int* __restrict__ idx_i) {
    int n = blockIdx.x * blockDim.x + threadIdx.x;
    if (n >= NVEC) return;
    float bv = bestv[n];
    int bi = besti[n];
    #pragma unroll
    for (int s = 1; s < NSEG; ++s) {        // ascending code ranges: strict '<'
        float v = bestv[(size_t)s * NVEC + n];   // keeps lowest index on ties
        int   i = besti[(size_t)s * NVEC + n];
        if (v < bv) { bv = v; bi = i; }
    }
    idx_i[n] = bi;
    idx_f[n] = (float)bi;
}

// -------- histogram of codes --------
__global__ void vq_counts_kernel(const int* __restrict__ idx_i, float* __restrict__ counts) {
    int n = blockIdx.x * blockDim.x + threadIdx.x;
    if (n < NVEC) atomicAdd(&counts[idx_i[n]], 1.0f);
}

// -------- dw scatter: dw[k,d] += z[n,d] --------
__global__ void vq_dw_kernel(const float* __restrict__ z, const int* __restrict__ idx_i,
                             float* __restrict__ dw) {
    int t = blockIdx.x * blockDim.x + threadIdx.x;   // coalesced over z (NCHW)
    if (t >= TOT) return;
    int b  = t / DHW;
    int d  = (t / HW) % DDIM;
    int hw = t % HW;
    int n  = b * HW + hw;
    atomicAdd(&dw[(size_t)idx_i[n] * DDIM + d], z[t]);
}

// -------- n = sum(cs*DECAY + (1-DECAY)*counts) (single block) --------
__global__ __launch_bounds__(1024)
void vq_reduce_n_kernel(const float* __restrict__ cs, const float* __restrict__ counts,
                        float* __restrict__ n_sum) {
    __shared__ float s[1024];
    float acc = 0.0f;
    for (int k = threadIdx.x; k < KCODES; k += 1024)
        acc += cs[k] * DECAY + ONE_M_DECAY * counts[k];
    s[threadIdx.x] = acc;
    __syncthreads();
    for (int off = 512; off > 0; off >>= 1) {
        if (threadIdx.x < off) s[threadIdx.x] += s[threadIdx.x + off];
        __syncthreads();
    }
    if (threadIdx.x == 0) n_sum[0] = s[0];
}

// -------- new_cs --------
__global__ void vq_cs_kernel(const float* __restrict__ cs, const float* __restrict__ counts,
                             const float* __restrict__ n_sum, float* __restrict__ out_cs) {
    int k = blockIdx.x * blockDim.x + threadIdx.x;
    if (k >= KCODES) return;
    float n = n_sum[0];
    float pre = cs[k] * DECAY + ONE_M_DECAY * counts[k];
    out_cs[k] = (pre + EPSV) / (n + (float)KCODES * EPSV) * n;
}

// -------- new_ema_w and new_embedding --------
__global__ void vq_w_kernel(const float* __restrict__ ema_w, const float* __restrict__ dw,
                            const float* __restrict__ new_cs,
                            float* __restrict__ out_emaw, float* __restrict__ out_emb) {
    int t = blockIdx.x * blockDim.x + threadIdx.x;
    if (t >= KCODES * DDIM) return;
    int k = t / DDIM;
    float ne = ema_w[t] * DECAY + ONE_M_DECAY * dw[t];
    out_emaw[t] = ne;
    out_emb[t] = ne / new_cs[k];
}

// -------- z_q (straight-through == embedding gather, NCHW) + loss --------
__global__ __launch_bounds__(256)
void vq_zq_loss_kernel(const float* __restrict__ z, const float* __restrict__ emb,
                       const int* __restrict__ idx_i,
                       float* __restrict__ out_zq, float* __restrict__ loss) {
    int t = blockIdx.x * blockDim.x + threadIdx.x;   // coalesced over z / out (NCHW)
    float diff2 = 0.0f;
    if (t < TOT) {
        int b  = t / DHW;
        int d  = (t / HW) % DDIM;
        int hw = t % HW;
        int n  = b * HW + hw;
        float e  = emb[(size_t)idx_i[n] * DDIM + d];
        float zv = z[t];
        out_zq[t] = e;                 // zp + sg(z_q - zp) == z_q numerically
        float df = e - zv;
        diff2 = df * df;
    }
    __shared__ float s[256];
    s[threadIdx.x] = diff2;
    __syncthreads();
    for (int off = 128; off > 0; off >>= 1) {
        if (threadIdx.x < off) s[threadIdx.x] += s[threadIdx.x + off];
        __syncthreads();
    }
    if (threadIdx.x == 0)
        atomicAdd(loss, s[0] * ((1.0f + BETA) / (float)TOT));
}

extern "C" void kernel_launch(void* const* d_in, const int* in_sizes, int n_in,
                              void* d_out, int out_size, void* d_ws, size_t ws_size,
                              hipStream_t stream) {
    const float* z      = (const float*)d_in[0];
    const float* emb    = (const float*)d_in[1];
    const float* cs     = (const float*)d_in[2];
    const float* ema_w  = (const float*)d_in[3];
    float* out = (float*)d_out;

    char* ws = (char*)d_ws;
    _Float16* zf16  = (_Float16*)(ws + WS_ZF16);
    _Float16* ef16  = (_Float16*)(ws + WS_EF16);
    float*    dw    = (float*)(ws + WS_DW);
    float*    enorm = (float*)(ws + WS_ENORM);
    float*    counts= (float*)(ws + WS_COUNTS);
    int*      idx_i = (int*)(ws + WS_IDXI);
    float*    n_sum = (float*)(ws + WS_NSUM);
    float*    bestv = (float*)(ws + WS_BESTV);
    int*      besti = (int*)(ws + WS_BESTI);

    const int B256 = 256;
    const int gKD  = (KCODES * DDIM + B256 - 1) / B256;   // 8192
    const int gTOT = (TOT + B256 - 1) / B256;             // 8192

    vq_init_kernel<<<gKD, B256, 0, stream>>>(dw, counts, n_sum, out + O_LOSS);
    vq_prep_z_kernel<<<gTOT, B256, 0, stream>>>(z, zf16);
    vq_prep_e_kernel<<<KCODES, B256, 0, stream>>>(emb, ef16, enorm);

    dim3 gWmma(NVEC / 32, NSEG);   // 256 x 4 = 1024 wave32 workgroups
    vq_argmin_wmma_kernel<<<gWmma, 32, 0, stream>>>(zf16, ef16, enorm, bestv, besti);
    vq_argmin_final_kernel<<<(NVEC + B256 - 1) / B256, B256, 0, stream>>>(
        bestv, besti, out + O_IDX, idx_i);

    vq_counts_kernel<<<(NVEC + B256 - 1) / B256, B256, 0, stream>>>(idx_i, counts);
    vq_dw_kernel<<<gTOT, B256, 0, stream>>>(z, idx_i, dw);
    vq_reduce_n_kernel<<<1, 1024, 0, stream>>>(cs, counts, n_sum);
    vq_cs_kernel<<<(KCODES + B256 - 1) / B256, B256, 0, stream>>>(cs, counts, n_sum,
                                                                  out + O_CS);
    vq_w_kernel<<<gKD, B256, 0, stream>>>(ema_w, dw, out + O_CS,
                                          out + O_EMAW, out + O_EMB);
    vq_zq_loss_kernel<<<gTOT, B256, 0, stream>>>(z, emb, idx_i,
                                                 out + O_ZQ, out + O_LOSS);
}
